// VectorQuantizerRestart_78529182040262
// MI455X (gfx1250) — matrix operand order repaired and checked by reading, not compiled, not run on hardware
//
#include <hip/hip_runtime.h>
#include <hip/hip_bf16.h>
#include <math.h>

typedef _Float16 half_t;
typedef __attribute__((ext_vector_type(16))) _Float16 v16h;
typedef __attribute__((ext_vector_type(8)))  _Float16 v8h;
typedef __attribute__((ext_vector_type(8)))  float    v8f;
typedef __attribute__((ext_vector_type(4)))  float    v4f;

#define N_E    1024
#define E_DIM  256
#define N_TOK  131072

#define CHUNK      64              // codes staged in LDS per iteration (32 KB f16)
#define WAVES      8               // waves per workgroup (wave32)
#define BLOCK      (WAVES * 32)
#define TOK_PER_WG (WAVES * 16)    // each wave owns a 16-token tile

// ---------------------------------------------------------------------------
// Kernel 1: weight f32 -> f16, per-code squared norms, zero the histogram.
// 1024 blocks x 32 threads (one wave per codebook row).
// ---------------------------------------------------------------------------
__global__ __launch_bounds__(32)
void vq_prep(const float* __restrict__ w, half_t* __restrict__ wf16,
             float* __restrict__ wnorm, unsigned int* __restrict__ counts) {
    const int code = blockIdx.x;
    const int lane = threadIdx.x;
    const v4f* row = (const v4f*)(w + (size_t)code * E_DIM);

    v4f a = row[lane * 2 + 0];
    v4f b = row[lane * 2 + 1];

    float s = 0.f;
    v8h hv;
#pragma unroll
    for (int j = 0; j < 4; ++j) {
        s += a[j] * a[j] + b[j] * b[j];
        hv[j]     = (half_t)a[j];
        hv[4 + j] = (half_t)b[j];
    }
    ((v8h*)(wf16 + (size_t)code * E_DIM))[lane] = hv;

#pragma unroll
    for (int m = 16; m >= 1; m >>= 1) s += __shfl_xor(s, m, 32);

    if (lane == 0) {
        wnorm[code]  = s;
        counts[code] = 0u;
    }
}

// ---------------------------------------------------------------------------
// Kernel 2: main VQ search. Per wave: 16-token tile, full K=256 A-fragments in
// VGPRs; codebook chunks staged in LDS; distances via v_wmma_f32_16x16x32_f16.
// ---------------------------------------------------------------------------
__global__ __launch_bounds__(BLOCK)
void vq_main(const float* __restrict__ z, const float* __restrict__ w,
             const half_t* __restrict__ wf16, const float* __restrict__ wnorm,
             unsigned int* __restrict__ counts, float* __restrict__ out) {
    __shared__ __align__(32) half_t ldsW[CHUNK * E_DIM];   // 32 KB: f16 code rows
    __shared__ float ldsN[CHUNK];                          // per-code ||e||^2
    __shared__ int   ldsIdx[TOK_PER_WG];                   // per-token argmin

    const int tid  = threadIdx.x;
    const int wave = tid >> 5;
    const int lane = tid & 31;
    const int l15  = lane & 15;
    const int h    = lane >> 4;          // 0: lanes 0-15, 1: lanes 16-31

    const int tokBase = blockIdx.x * TOK_PER_WG + wave * 16;

    // ---- Preload A fragments (this wave's 16 tokens, all of K=0..255) ----
    // A layout (matches B): lane<16 holds K = kb..kb+15 for row M=lane;
    // lanes 16-31 hold K = kb+16..kb+31 for M=lane-16.
    v16h afrag[8];
    {
        const float* zp = z + (size_t)(tokBase + l15) * E_DIM;
#pragma unroll
        for (int kf = 0; kf < 8; ++kf) {
            const int kb = kf * 32;
            const int g0 = kb + 16 * h;       // first 8-wide K group
            const int g1 = kb + 16 * h + 8;   // second 8-wide K group
            v4f a0 = *(const v4f*)(zp + g0);
            v4f a1 = *(const v4f*)(zp + g0 + 4);
            v4f b0 = *(const v4f*)(zp + g1);
            v4f b1 = *(const v4f*)(zp + g1 + 4);
            v16h f;
#pragma unroll
            for (int j = 0; j < 4; ++j) {
                f[j]      = (half_t)a0[j];
                f[4 + j]  = (half_t)a1[j];
                f[8 + j]  = (half_t)b0[j];
                f[12 + j] = (half_t)b1[j];
            }
            afrag[kf] = f;
        }
    }

    // Running argmin state: VGPR r of the C tile maps to token row (r + 8*h).
    float bestVal[8];
    int   bestIdx[8];
#pragma unroll
    for (int r = 0; r < 8; ++r) { bestVal[r] = 3.4e38f; bestIdx[r] = 0; }

    // ---- Sweep codebook in LDS-staged chunks ----
    for (int cb = 0; cb < N_E; cb += CHUNK) {
        __syncthreads();
        {   // stage CHUNK f16 code rows: 32768 B, 256 threads x 8 x b128
            const uint4* src = (const uint4*)(wf16 + (size_t)cb * E_DIM);
            uint4*       dst = (uint4*)ldsW;
#pragma unroll
            for (int i = tid; i < CHUNK * E_DIM / 8; i += BLOCK) dst[i] = src[i];
            if (tid < CHUNK) ldsN[tid] = wnorm[cb + tid];
        }
        __syncthreads();

        // prefetch next chunk into cache while we compute on this one
        if (cb + CHUNK < N_E)
            __builtin_prefetch(wf16 + (size_t)(cb + CHUNK) * E_DIM + tid * 64, 0, 0);

#pragma unroll
        for (int st = 0; st < CHUNK / 16; ++st) {
            v8f acc = {};
            // B layout used here (32x16, same k order as A): lane<16 holds K=kb..kb+15 of
            // column N=lane; lanes 16-31 hold K=kb+16..kb+31 of column N=lane-16.
            const half_t* wl = ldsW + (size_t)(st * 16 + l15) * E_DIM + 16 * h;
#pragma unroll
            for (int kf = 0; kf < 8; ++kf) {
                v16h bfrag = *(const v16h*)(wl + kf * 32);
                acc = __builtin_amdgcn_wmma_f32_16x16x32_f16(
                        false, afrag[kf], false, bfrag, (short)0, acc, false, false);
            }
            const float wn = ldsN[st * 16 + l15];
            const int   n  = cb + st * 16 + l15;   // code column this lane sees
#pragma unroll
            for (int r = 0; r < 8; ++r) {
                const float score = wn - 2.0f * acc[r];  // ||e||^2 - 2 z.e
                if (score < bestVal[r]) { bestVal[r] = score; bestIdx[r] = n; }
            }
        }
    }

    // ---- Argmin reduction across the 16 code-columns (within each half) ----
#pragma unroll
    for (int r = 0; r < 8; ++r) {
        float v  = bestVal[r];
        int   ix = bestIdx[r];
#pragma unroll
        for (int m = 8; m >= 1; m >>= 1) {
            float ov = __shfl_xor(v, m, 32);
            int   oi = __shfl_xor(ix, m, 32);
            if (ov < v || (ov == v && oi < ix)) { v = ov; ix = oi; }  // first-min tie
        }
        if (l15 == 0) ldsIdx[wave * 16 + r + 8 * h] = ix;
    }
    __syncthreads();

    // ---- Histogram: one atomic per token ----
    if (lane < 16) atomicAdd(&counts[ldsIdx[wave * 16 + lane]], 1u);

    // ---- z_q = weight[idx] gather (f32, wave-cooperative row copies) ----
#pragma unroll 4
    for (int t = 0; t < 16; ++t) {
        const int ix = ldsIdx[wave * 16 + t];
        const v4f* src = (const v4f*)(w + (size_t)ix * E_DIM);
        v4f* dst = (v4f*)(out + (size_t)(tokBase + t) * E_DIM);
        dst[lane]      = src[lane];        // 64 x float4 per row, 32 lanes x 2
        dst[lane + 32] = src[lane + 32];
    }
}

// ---------------------------------------------------------------------------
// Kernel 3: perplexity from the histogram. One block.
// ---------------------------------------------------------------------------
__global__ __launch_bounds__(256)
void vq_perp(const unsigned int* __restrict__ counts, float* __restrict__ out) {
    __shared__ float red[8];
    float s = 0.f;
    for (int i = threadIdx.x; i < N_E; i += 256) {
        const float em = (float)counts[i] * (1.0f / (float)N_TOK);
        s += em * logf(em + 1e-10f);
    }
#pragma unroll
    for (int m = 16; m >= 1; m >>= 1) s += __shfl_xor(s, m, 32);
    const int wv = threadIdx.x >> 5;
    if ((threadIdx.x & 31) == 0) red[wv] = s;
    __syncthreads();
    if (threadIdx.x == 0) {
        float t = 0.f;
#pragma unroll
        for (int i = 0; i < 8; ++i) t += red[i];
        out[(size_t)N_TOK * E_DIM] = expf(-t);
    }
}

// ---------------------------------------------------------------------------
extern "C" void kernel_launch(void* const* d_in, const int* in_sizes, int n_in,
                              void* d_out, int out_size, void* d_ws, size_t ws_size,
                              hipStream_t stream) {
    const float* z = (const float*)d_in[0];   // [131072, 256] f32
    const float* w = (const float*)d_in[1];   // [1024, 256]   f32
    float* out = (float*)d_out;               // z_q flat (N*D) + perplexity (1)

    // workspace layout
    half_t*       wf16   = (half_t*)d_ws;                                   // 512 KB
    float*        wnorm  = (float*)((char*)d_ws + (size_t)N_E * E_DIM * 2); //   4 KB
    unsigned int* counts = (unsigned int*)((char*)wnorm + N_E * sizeof(float));

    vq_prep<<<N_E, 32, 0, stream>>>(w, wf16, wnorm, counts);
    vq_main<<<N_TOK / TOK_PER_WG, BLOCK, 0, stream>>>(z, w, wf16, wnorm, counts, out);
    vq_perp<<<1, 256, 0, stream>>>(counts, out);
}
